// attnGCN_81363860455712
// MI455X (gfx1250) — matrix operand order repaired and checked by reading, not compiled, not run on hardware
//
#include <hip/hip_runtime.h>
#include <hip/hip_bf16.h>
#include <math.h>

typedef __attribute__((ext_vector_type(16))) _Float16 v16h;
typedef __attribute__((ext_vector_type(8)))  _Float16 v8h;
typedef __attribute__((ext_vector_type(8)))  float    v8f;

#define F_DIM 128
#define H_DIM 128
#define NGRAPH 64

// ---------------------------------------------------------------------------
// WMMA fragment loaders (gfx1250 wave32 layouts, CDNA5 ISA §7.12.2)
// A (16x32 f16): lane m<16 holds row m, K = {k0+0..7, k0+16..23};
//                lane m+16 holds row m, K = {k0+8..15, k0+24..31}
// B (32x16 f16): lane c<16 holds column c, K = {k0+0..7, k0+16..23};
//                lane c+16 holds column c, K = {k0+8..15, k0+24..31}.
// B is supplied transposed (Bt[n][k], row-major, contiguous in K).
// ---------------------------------------------------------------------------
__device__ __forceinline__ v16h load_frag_a(const _Float16* __restrict__ base,
                                            int row0, int K, int k0, int lm, int lh) {
  const _Float16* p = base + (long long)(row0 + lm) * K + k0 + lh * 8;
  v8h lo = *(const v8h*)(p);
  v8h hi = *(const v8h*)(p + 16);
  v16h f;
#pragma unroll
  for (int t = 0; t < 8; ++t) { f[t] = lo[t]; f[t + 8] = hi[t]; }
  return f;
}

__device__ __forceinline__ v16h load_frag_b(const _Float16* __restrict__ base,
                                            int row0, int K, int k0, int lm, int lh) {
  const _Float16* p = base + (long long)(row0 + lm) * K + k0 + lh * 8;
  v8h lo = *(const v8h*)(p);
  v8h hi = *(const v8h*)(p + 16);
  v16h f;
#pragma unroll
  for (int t = 0; t < 8; ++t) { f[t] = lo[t]; f[t + 8] = hi[t]; }
  return f;
}

// ---------------------------------------------------------------------------
// Batched NT GEMM: C[b] = scale * A[b] (MxK, row-major f16) * Bt[b]^T
//   Bt given as (NxK, row-major f16)  => C is MxN.
// Block: 256 threads = 8 waves, block tile 128(M) x 64(N),
// each wave computes a 32x32 tile as 2x2 v_wmma_f32_16x16x32_f16 fragments.
// store_half: 0 -> f32 C, 1 -> f16 C.
// Assumes M%128==0, N%64==0, K%32==0 (true for all uses here).
// ---------------------------------------------------------------------------
__global__ __launch_bounds__(256) void gemm_nt_wmma(
    const _Float16* __restrict__ A, const _Float16* __restrict__ Bt,
    void* __restrict__ Cout, int M, int N, int K,
    long long sA, long long sB, long long sC, float scale, int store_half)
{
  const _Float16* Ab = A  + (long long)blockIdx.z * sA;
  const _Float16* Bb = Bt + (long long)blockIdx.z * sB;
  const int lane = threadIdx.x & 31;
  const int wave = threadIdx.x >> 5;
  const int wm = wave >> 1;          // 0..3
  const int wn = wave & 1;           // 0..1
  const int m0 = blockIdx.y * 128 + wm * 32;
  const int n0 = blockIdx.x * 64  + wn * 32;
  const int lm = lane & 15;
  const int lh = lane >> 4;

  v8f acc[2][2] = {};
  for (int k0 = 0; k0 < K; k0 += 32) {
    v16h af[2], bf[2];
#pragma unroll
    for (int i = 0; i < 2; ++i) af[i] = load_frag_a(Ab, m0 + i * 16, K, k0, lm, lh);
#pragma unroll
    for (int j = 0; j < 2; ++j) bf[j] = load_frag_b(Bb, n0 + j * 16, K, k0, lm, lh);
#pragma unroll
    for (int i = 0; i < 2; ++i)
#pragma unroll
      for (int j = 0; j < 2; ++j)
        acc[i][j] = __builtin_amdgcn_wmma_f32_16x16x32_f16(
            false, af[i], false, bf[j], (short)0, acc[i][j], false, false);
  }

  // C/D layout: lane<16 -> (M = lh*8 + v, N = lm); lane>=16 -> M += 8.
#pragma unroll
  for (int i = 0; i < 2; ++i) {
#pragma unroll
    for (int j = 0; j < 2; ++j) {
      const int row0 = m0 + i * 16 + lh * 8;
      const int col  = n0 + j * 16 + lm;
      const long long base = (long long)row0 * N + col;
      if (store_half) {
        _Float16* C = (_Float16*)Cout + (long long)blockIdx.z * sC;
#pragma unroll
        for (int v = 0; v < 8; ++v)
          C[base + (long long)v * N] = (_Float16)(acc[i][j][v] * scale);
      } else {
        float* C = (float*)Cout + (long long)blockIdx.z * sC;
#pragma unroll
        for (int v = 0; v < 8; ++v)
          C[base + (long long)v * N] = acc[i][j][v] * scale;
      }
    }
  }
}

// ---------------------------------------------------------------------------
// Elementwise / graph kernels
// ---------------------------------------------------------------------------
__global__ void wtrans_k(const float* __restrict__ W, _Float16* __restrict__ Wt) {
  int idx = blockIdx.x * 256 + threadIdx.x;      // 128*128
  int n = idx >> 7, k = idx & 127;
  Wt[idx] = (_Float16)W[k * 128 + n];
}

__global__ void cvt_f16_k(const float* __restrict__ s, _Float16* __restrict__ d, long long total) {
  long long i = (long long)blockIdx.x * 256 + threadIdx.x;
  if (i < total) d[i] = (_Float16)s[i];
}

__global__ void deg_init_k(float* deg, int Nn) {
  int i = blockIdx.x * 256 + threadIdx.x;
  if (i < Nn) deg[i] = 1.0f;                      // self-loop
}

__global__ void deg_acc_k(const int* __restrict__ ei, float* deg, int E) {
  int e = blockIdx.x * 256 + threadIdx.x;
  if (e < E) atomicAdd(&deg[ei[E + e]], 1.0f);    // row 1 of edge_index = dst
}

__global__ void dinv_k(const float* __restrict__ deg, float* dinv, int Nn) {
  int i = blockIdx.x * 256 + threadIdx.x;
  if (i < Nn) dinv[i] = rsqrtf(fmaxf(deg[i], 1.0f));
}

__global__ void init_acc_k(const float* __restrict__ t, const float* __restrict__ dinv,
                           float* __restrict__ acc, long long total) {
  long long idx = (long long)blockIdx.x * 256 + threadIdx.x;
  if (idx >= total) return;
  int i = (int)(idx >> 7);
  float dv = dinv[i];
  acc[idx] = t[idx] * dv * dv;                    // self-loop contribution
}

__global__ void scatter_k(const int* __restrict__ ei, const float* __restrict__ t,
                          const float* __restrict__ dinv, float* __restrict__ acc, int E) {
  long long idx = (long long)blockIdx.x * 256 + threadIdx.x;
  if (idx >= (long long)E * 128) return;
  int e = (int)(idx >> 7), f = (int)(idx & 127);
  int s = ei[e], d = ei[E + e];
  atomicAdd(&acc[(long long)d * 128 + f], t[(long long)s * 128 + f] * dinv[s] * dinv[d]);
}

__global__ void finalize_k(const float* __restrict__ acc, const float* __restrict__ bias,
                           _Float16* __restrict__ h16, long long total) {
  long long idx = (long long)blockIdx.x * 256 + threadIdx.x;
  if (idx >= total) return;
  int f = (int)(idx & 127);
  h16[idx] = (_Float16)fmaxf(acc[idx] + bias[f], 0.0f);
}

__global__ void bias_cvt_k(const float* __restrict__ t, const float* __restrict__ bias,
                           _Float16* __restrict__ o, long long total) {
  long long idx = (long long)blockIdx.x * 256 + threadIdx.x;
  if (idx >= total) return;
  int f = (int)(idx & 127);
  o[idx] = (_Float16)(t[idx] + bias[f]);
}

// vt[g][f][j] = f16(v[g*n + j][f] + bv[f])   (n = 512 nodes per graph)
__global__ void vtrans_k(const float* __restrict__ t, const float* __restrict__ bv,
                         _Float16* __restrict__ vt, long long total) {
  long long idx = (long long)blockIdx.x * 256 + threadIdx.x;
  if (idx >= total) return;
  int f = (int)(idx & 127);
  int node = (int)(idx >> 7);
  int g = node >> 9, j = node & 511;
  vt[((long long)g * 128 + f) * 512 + j] = (_Float16)(t[idx] + bv[f]);
}

// In-place row softmax on f16 scores; one block (256 thr) per row of `cols`.
__global__ void softmax_rows_f16(_Float16* __restrict__ S, int cols) {
  long long row = blockIdx.x;
  _Float16* p = S + row * (long long)cols;
  __shared__ float red[256];
  int tid = threadIdx.x;
  float mx = -1e30f;
  for (int c = tid; c < cols; c += 256) mx = fmaxf(mx, (float)p[c]);
  red[tid] = mx; __syncthreads();
  for (int s = 128; s > 0; s >>= 1) { if (tid < s) red[tid] = fmaxf(red[tid], red[tid + s]); __syncthreads(); }
  mx = red[0]; __syncthreads();
  float sum = 0.0f;
  for (int c = tid; c < cols; c += 256) sum += __expf((float)p[c] - mx);
  red[tid] = sum; __syncthreads();
  for (int s = 128; s > 0; s >>= 1) { if (tid < s) red[tid] += red[tid + s]; __syncthreads(); }
  float inv = 1.0f / red[0]; __syncthreads();
  for (int c = tid; c < cols; c += 256) p[c] = (_Float16)(__expf((float)p[c] - mx) * inv);
}

__global__ void mean_pool_k(const float* __restrict__ O, float* __restrict__ pooled, int n) {
  int g = blockIdx.x, f = threadIdx.x;           // 64 blocks x 128 threads
  const float* p = O + (long long)g * n * 128 + f;
  float s = 0.0f;
  for (int j = 0; j < n; ++j) s += p[(long long)j * 128];
  pooled[g * 128 + f] = s * (1.0f / (float)n);
}

__global__ void head_k(const float* __restrict__ pooled, const float* __restrict__ Wh1,
                       const float* __restrict__ bh1, const float* __restrict__ Wh2,
                       const float* __restrict__ bh2, float* __restrict__ out) {
  int g = blockIdx.x, m = threadIdx.x;           // 64 blocks x 64 threads
  const float* pg = pooled + g * 128;
  float s = bh1[m];
  for (int f = 0; f < 128; ++f) s += pg[f] * Wh1[f * 64 + m];
  float v = fmaxf(s, 0.0f) * Wh2[m];
  __shared__ float red[64];
  red[m] = v; __syncthreads();
  for (int st = 32; st > 0; st >>= 1) { if (m < st) red[m] += red[m + st]; __syncthreads(); }
  if (m == 0) out[g] = red[0] + bh2[0];
}

// ---------------------------------------------------------------------------
extern "C" void kernel_launch(void* const* d_in, const int* in_sizes, int n_in,
                              void* d_out, int out_size, void* d_ws, size_t ws_size,
                              hipStream_t stream)
{
  const float* x   = (const float*)d_in[0];
  const int*   ei  = (const int*)  d_in[1];
  const float* W1  = (const float*)d_in[3];
  const float* b1  = (const float*)d_in[4];
  const float* W2  = (const float*)d_in[5];
  const float* b2  = (const float*)d_in[6];
  const float* Wq  = (const float*)d_in[7];
  const float* bq  = (const float*)d_in[8];
  const float* Wk  = (const float*)d_in[9];
  const float* bk  = (const float*)d_in[10];
  const float* Wv  = (const float*)d_in[11];
  const float* bv  = (const float*)d_in[12];
  const float* Wh1 = (const float*)d_in[13];
  const float* bh1 = (const float*)d_in[14];
  const float* Wh2 = (const float*)d_in[15];
  const float* bh2 = (const float*)d_in[16];
  float* out = (float*)d_out;

  const int Nn = in_sizes[0] / F_DIM;   // 32768
  const int E  = in_sizes[1] / 2;       // 524288
  const int n  = Nn / NGRAPH;           // 512

  // ---- workspace layout -------------------------------------------------
  char* w = (char*)d_ws;
  auto alloc = [&](size_t bytes) -> char* {
    char* p = w; w += (bytes + 255) & ~(size_t)255; return p;
  };
  _Float16* xf16 = (_Float16*)alloc((size_t)Nn * F_DIM * 2);
  _Float16* h16  = (_Float16*)alloc((size_t)Nn * H_DIM * 2);
  float*    t    = (float*)   alloc((size_t)Nn * H_DIM * 4);
  float*    accb = (float*)   alloc((size_t)Nn * H_DIM * 4);
  float*    deg  = (float*)   alloc((size_t)Nn * 4);
  float*    dinv = (float*)   alloc((size_t)Nn * 4);
  _Float16* W1t  = (_Float16*)alloc(128 * 128 * 2);
  _Float16* W2t  = (_Float16*)alloc(128 * 128 * 2);
  _Float16* Wqt  = (_Float16*)alloc(128 * 128 * 2);
  _Float16* Wkt  = (_Float16*)alloc(128 * 128 * 2);
  _Float16* Wvt  = (_Float16*)alloc(128 * 128 * 2);
  _Float16* q16  = (_Float16*)alloc((size_t)Nn * H_DIM * 2);
  _Float16* k16  = (_Float16*)alloc((size_t)Nn * H_DIM * 2);
  _Float16* vt16 = (_Float16*)alloc((size_t)Nn * H_DIM * 2);
  _Float16* S    = (_Float16*)alloc((size_t)NGRAPH * n * n * 2);
  float*   pooled= (float*)   alloc((size_t)NGRAPH * H_DIM * 4);
  float*   O     = t;  // reuse: t is free when P@V runs

  const long long totNH = (long long)Nn * H_DIM;           // 4,194,304
  const int ebNH = (int)((totNH + 255) / 256);
  const long long totEF = (long long)E * 128;
  const int ebEF = (int)((totEF + 255) / 256);
  const long long sNode = totNH;                            // per-graph strides
  const long long sQK   = (long long)n * H_DIM;             // 65536
  const long long sSS   = (long long)n * n;                 // 262144

  // ---- weight prep & input convert --------------------------------------
  wtrans_k<<<64, 256, 0, stream>>>(W1, W1t);
  wtrans_k<<<64, 256, 0, stream>>>(W2, W2t);
  wtrans_k<<<64, 256, 0, stream>>>(Wq, Wqt);
  wtrans_k<<<64, 256, 0, stream>>>(Wk, Wkt);
  wtrans_k<<<64, 256, 0, stream>>>(Wv, Wvt);
  cvt_f16_k<<<ebNH, 256, 0, stream>>>(x, xf16, totNH);

  // ---- degrees ----------------------------------------------------------
  deg_init_k<<<(Nn + 255) / 256, 256, 0, stream>>>(deg, Nn);
  deg_acc_k<<<(E + 255) / 256, 256, 0, stream>>>(ei, deg, E);
  dinv_k<<<(Nn + 255) / 256, 256, 0, stream>>>(deg, dinv, Nn);

  dim3 gNode(H_DIM / 64, Nn / 128, 1);

  // ---- GCN layer 1 ------------------------------------------------------
  gemm_nt_wmma<<<gNode, 256, 0, stream>>>(xf16, W1t, t, Nn, H_DIM, F_DIM,
                                          0, 0, 0, 1.0f, 0);
  init_acc_k<<<ebNH, 256, 0, stream>>>(t, dinv, accb, totNH);
  scatter_k<<<ebEF, 256, 0, stream>>>(ei, t, dinv, accb, E);
  finalize_k<<<ebNH, 256, 0, stream>>>(accb, b1, h16, totNH);

  // ---- GCN layer 2 ------------------------------------------------------
  gemm_nt_wmma<<<gNode, 256, 0, stream>>>(h16, W2t, t, Nn, H_DIM, H_DIM,
                                          0, 0, 0, 1.0f, 0);
  init_acc_k<<<ebNH, 256, 0, stream>>>(t, dinv, accb, totNH);
  scatter_k<<<ebEF, 256, 0, stream>>>(ei, t, dinv, accb, E);
  finalize_k<<<ebNH, 256, 0, stream>>>(accb, b2, h16, totNH);

  // ---- Q / K / V --------------------------------------------------------
  gemm_nt_wmma<<<gNode, 256, 0, stream>>>(h16, Wqt, t, Nn, H_DIM, H_DIM,
                                          0, 0, 0, 1.0f, 0);
  bias_cvt_k<<<ebNH, 256, 0, stream>>>(t, bq, q16, totNH);
  gemm_nt_wmma<<<gNode, 256, 0, stream>>>(h16, Wkt, t, Nn, H_DIM, H_DIM,
                                          0, 0, 0, 1.0f, 0);
  bias_cvt_k<<<ebNH, 256, 0, stream>>>(t, bk, k16, totNH);
  gemm_nt_wmma<<<gNode, 256, 0, stream>>>(h16, Wvt, t, Nn, H_DIM, H_DIM,
                                          0, 0, 0, 1.0f, 0);
  vtrans_k<<<ebNH, 256, 0, stream>>>(t, bv, vt16, totNH);

  // ---- attention scores: S[g] = q[g] @ k[g]^T / sqrt(H), f16 out --------
  dim3 gScore(n / 64, n / 128, NGRAPH);
  gemm_nt_wmma<<<gScore, 256, 0, stream>>>(q16, k16, S, n, n, H_DIM,
                                           sQK, sQK, sSS,
                                           0.08838834764831845f /*1/sqrt(128)*/, 1);

  // ---- softmax (in place, per row) --------------------------------------
  softmax_rows_f16<<<NGRAPH * n, 256, 0, stream>>>(S, n);

  // ---- O[g] = P[g] @ V[g]  (NT against vt) ------------------------------
  dim3 gPV(H_DIM / 64, n / 128, NGRAPH);
  gemm_nt_wmma<<<gPV, 256, 0, stream>>>(S, vt16, O, n, H_DIM, n,
                                        sSS, sQK, sQK, 1.0f, 0);

  // ---- mean pool + MLP head --------------------------------------------
  mean_pool_k<<<NGRAPH, 128, 0, stream>>>(O, pooled, n);
  head_k<<<NGRAPH, 64, 0, stream>>>(pooled, Wh1, bh1, Wh2, bh2, out);

  (void)n_in; (void)out_size; (void)ws_size; (void)sNode;
}
